// DeepSeekV3TokenChoiceTopKRouter_19550691131496
// MI455X (gfx1250) — matrix operand order repaired and checked by reading, not compiled, not run on hardware
//
#include <hip/hip_runtime.h>
#include <hip/hip_bf16.h>

#define NUM_EXPERTS 256
#define DIM 7168
#define N_TOKENS 8192
#define M_TILE 64          // tokens per block (chunk) -- halves gate_w L2 traffic vs 32
#define K_STEP 32
#define THREADS 512        // 16 waves: 4 M-stripes x 4 N-quarters
#define ROW_BYTES 80       // (32 data + 8 pad) bf16 * 2B; 16B aligned; 20-dword stride (conflict-free)
#define N_CHUNKS (N_TOKENS / M_TILE)   // 128
#define CHUNK_ENTRIES (M_TILE * 8)     // 512

typedef __attribute__((ext_vector_type(16))) __bf16 v16bf;
typedef __attribute__((ext_vector_type(8)))  float  v8f;

union FragBF { v16bf v; uint4 q[2]; };

// Dynamic LDS layout (GEMM phase):
//   [    0,  5120) A_hi   64 rows x 80B
//   [ 5120, 10240) A_lo
//   [10240, 30720) B_hi^T 256 rows x 80B
//   [30720, 51200) B_lo^T
// Overlay (routing phase, tiles dead; two passes of 32 tokens):
//   [    0, 32768) scores 32 x 256 f32
//   [32768, 33792) hist   256 int (persists across both passes)
#define SM_AHI    0
#define SM_ALO    5120
#define SM_BHI    10240
#define SM_BLO    30720
#define SM_BYTES  51200
#define SM_SCORES 0
#define SM_HIST   32768

__device__ __forceinline__ unsigned short bf_bits(__bf16 h) {
  return __builtin_bit_cast(unsigned short, h);
}
__device__ __forceinline__ unsigned pack2(__bf16 a, __bf16 b) {
  return (unsigned)bf_bits(a) | ((unsigned)bf_bits(b) << 16);
}
__device__ __forceinline__ void split_bf(float f, __bf16& hi, __bf16& lo) {
  hi = (__bf16)f;
  lo = (__bf16)(f - (float)hi);
}

__global__ __launch_bounds__(THREADS)
void moe_gemm_router_kernel(const float* __restrict__ x,
                            const float* __restrict__ gw,
                            const float* __restrict__ bias,
                            int*   __restrict__ topk_ids,
                            float* __restrict__ topk_w,
                            int*   __restrict__ chunk_counts) {
  extern __shared__ char sm[];
  const int t    = threadIdx.x;
  const int lane = t & 31;
  const int w    = t >> 5;        // 0..15
  const int lm   = lane & 15;
  const int sel  = lane >> 4;
  const int waveM = w & 3;        // 4 M-stripes of 16 tokens
  const int waveN = w >> 2;       // 4 N-quarters of 64 experts
  const int mBase = blockIdx.x * M_TILE;

  const v8f vzero = {0.f, 0.f, 0.f, 0.f, 0.f, 0.f, 0.f, 0.f};
  v8f acc[4];
  #pragma unroll
  for (int j = 0; j < 4; ++j) acc[j] = vzero;

  float biasReg[4];
  #pragma unroll
  for (int j = 0; j < 4; ++j) biasReg[j] = bias[waveN * 64 + j * 16 + lm];

  // Cooperative-load geometry
  const int am = t >> 3;          // 0..63 : token row of A tile
  const int ak = (t & 7) * 4;     // 0,4,..,28 : k (float4)
  const float* xrow = x + (size_t)(mBase + am) * DIM + ak;
  const int be_ = t & 255;        // expert column owned for B staging
  const int bkh = t >> 8;         // 0/1 : which half of the 32-k slab
  const float* wcol = gw + be_;

  for (int k0 = 0; k0 < DIM; k0 += K_STEP) {
    if (k0 + K_STEP < DIM) {
      __builtin_prefetch(xrow + k0 + K_STEP, 0, 1);
      __builtin_prefetch(wcol + (size_t)(k0 + K_STEP + bkh * 16) * NUM_EXPERTS, 0, 1);
    }
    // ---- A tile: fp32 -> split bf16, row-major in LDS ----
    float4 xv = *(const float4*)(xrow + k0);
    __bf16 h0, h1, h2, h3, l0, l1, l2, l3;
    split_bf(xv.x, h0, l0); split_bf(xv.y, h1, l1);
    split_bf(xv.z, h2, l2); split_bf(xv.w, h3, l3);
    *(uint2*)(sm + SM_AHI + am * ROW_BYTES + ak * 2) = make_uint2(pack2(h0, h1), pack2(h2, h3));
    *(uint2*)(sm + SM_ALO + am * ROW_BYTES + ak * 2) = make_uint2(pack2(l0, l1), pack2(l2, l3));
    // ---- B tile: fp32 -> split bf16, transposed [expert][k] in LDS ----
    unsigned hprev = 0, lprev = 0;
    #pragma unroll
    for (int kk = 0; kk < 16; ++kk) {
      const int kkk = bkh * 16 + kk;
      float f = wcol[(size_t)(k0 + kkk) * NUM_EXPERTS];
      __bf16 bh, bl; split_bf(f, bh, bl);
      if (kkk & 1) {
        *(unsigned*)(sm + SM_BHI + be_ * ROW_BYTES + (kkk - 1) * 2) = hprev | ((unsigned)bf_bits(bh) << 16);
        *(unsigned*)(sm + SM_BLO + be_ * ROW_BYTES + (kkk - 1) * 2) = lprev | ((unsigned)bf_bits(bl) << 16);
      } else {
        hprev = bf_bits(bh); lprev = bf_bits(bl);
      }
    }
    __syncthreads();

    // ---- fragments ----
    // A 16x32 bf16 (ISA layout): sel=0 -> K0..7 / K16..23 ; sel=1 -> K8..15 / K24..31
    const int aRow = (waveM * 16 + lm) * ROW_BYTES + sel * 16;
    FragBF aHi, aLo;
    aHi.q[0] = *(const uint4*)(sm + SM_AHI + aRow);
    aHi.q[1] = *(const uint4*)(sm + SM_AHI + aRow + 32);
    aLo.q[0] = *(const uint4*)(sm + SM_ALO + aRow);
    aLo.q[1] = *(const uint4*)(sm + SM_ALO + aRow + 32);
    #pragma unroll
    for (int j = 0; j < 4; ++j) {
      // B 32x16 bf16 (ISA layout): sel=0 -> K0..7 / K16..23 ; sel=1 -> K8..15 / K24..31
      const int eRow = (waveN * 64 + j * 16 + lm) * ROW_BYTES + sel * 16;
      FragBF bHi, bLo;
      bHi.q[0] = *(const uint4*)(sm + SM_BHI + eRow);
      bHi.q[1] = *(const uint4*)(sm + SM_BHI + eRow + 32);
      bLo.q[0] = *(const uint4*)(sm + SM_BLO + eRow);
      bLo.q[1] = *(const uint4*)(sm + SM_BLO + eRow + 32);
      // fp32-accurate GEMM via bf16 split: hi*hi + lo*hi + hi*lo
      acc[j] = __builtin_amdgcn_wmma_f32_16x16x32_bf16(false, aHi.v, false, bHi.v, (short)0, acc[j], false, false);
      acc[j] = __builtin_amdgcn_wmma_f32_16x16x32_bf16(false, aLo.v, false, bHi.v, (short)0, acc[j], false, false);
      acc[j] = __builtin_amdgcn_wmma_f32_16x16x32_bf16(false, aHi.v, false, bLo.v, (short)0, acc[j], false, false);
    }
    __syncthreads();
  }

  // ---- routing phase: two passes of 32 tokens over a 32KB score overlay ----
  float* scores = (float*)(sm + SM_SCORES);
  int*   hist   = (int*)(sm + SM_HIST);
  if (t < NUM_EXPERTS) hist[t] = 0;

  const float NEG_INF = -__builtin_inff();
  #pragma unroll
  for (int p = 0; p < 2; ++p) {
    // waves whose M-stripe falls in this pass write sigmoid+bias scores
    if ((waveM >> 1) == p) {
      #pragma unroll
      for (int j = 0; j < 4; ++j) {
        const int e = waveN * 64 + j * 16 + lm;
        #pragma unroll
        for (int r = 0; r < 8; ++r) {
          const int tr = (waveM & 1) * 16 + r + sel * 8;   // token within pass (C/D layout: lanes16-31 = M+8)
          float s = 1.f / (1.f + __expf(-acc[j][r])) + biasReg[j];
          scores[tr * NUM_EXPERTS + e] = s;
        }
      }
    }
    __syncthreads();

    // each of the 16 waves routes 2 tokens; lane l owns experts l+32j
    for (int i = 0; i < 2; ++i) {
      const int tr  = w * 2 + i;               // 0..31 within pass
      const int gt  = mBase + p * 32 + tr;     // global token
      const float* row = scores + tr * NUM_EXPERTS;
      float s[8];
      #pragma unroll
      for (int j = 0; j < 8; ++j) s[j] = row[lane + 32 * j];

      // group score = sum of top-2 within each group of 32 lanes
      float gsum[8];
      #pragma unroll
      for (int j = 0; j < 8; ++j) {
        float v = s[j]; int li = lane;
        #pragma unroll
        for (int off = 16; off > 0; off >>= 1) {
          float ov = __shfl_xor(v, off, 32);
          int   ol = __shfl_xor(li, off, 32);
          if (ov > v || (ov == v && ol < li)) { v = ov; li = ol; }
        }
        float v2 = (lane == li) ? NEG_INF : s[j];
        #pragma unroll
        for (int off = 16; off > 0; off >>= 1) {
          float ov = __shfl_xor(v2, off, 32);
          v2 = (ov > v2) ? ov : v2;
        }
        gsum[j] = v + v2;
      }
      // top-4 groups (ties -> lower group index, matches lax.top_k)
      unsigned gmask = 0;
      #pragma unroll
      for (int r4 = 0; r4 < 4; ++r4) {
        float bv = NEG_INF; int bg = 0;
        #pragma unroll
        for (int j = 0; j < 8; ++j)
          if (!((gmask >> j) & 1) && gsum[j] > bv) { bv = gsum[j]; bg = j; }
        gmask |= (1u << bg);
      }
      float ms[8];
      #pragma unroll
      for (int j = 0; j < 8; ++j) ms[j] = ((gmask >> j) & 1) ? s[j] : NEG_INF;

      // top-8 experts among masked (ties -> lower expert index)
      float sum = 0.f; int myE = 0; float myV = 0.f;
      #pragma unroll
      for (int k = 0; k < 8; ++k) {
        float bv = NEG_INF; int bei = 0x7fffffff;
        #pragma unroll
        for (int j = 0; j < 8; ++j) {
          int e = lane + 32 * j;
          if (ms[j] > bv || (ms[j] == bv && e < bei)) { bv = ms[j]; bei = e; }
        }
        #pragma unroll
        for (int off = 16; off > 0; off >>= 1) {
          float ov = __shfl_xor(bv, off, 32);
          int   oe = __shfl_xor(bei, off, 32);
          if (ov > bv || (ov == bv && oe < bei)) { bv = ov; bei = oe; }
        }
        sum += bv;
        if (lane == (bei & 31)) {
          const int jj = bei >> 5;
          #pragma unroll
          for (int j = 0; j < 8; ++j) if (j == jj) ms[j] = NEG_INF;
        }
        if (lane == k) { myE = bei; myV = bv; }
      }
      const float scale = 2.5f / (sum + 1e-20f);
      if (lane < 8) {
        topk_ids[gt * 8 + lane] = myE;
        topk_w[gt * 8 + lane]   = myV * scale;
        atomicAdd(&hist[myE], 1);
      }
    }
    __syncthreads();   // protect score overlay before next pass / final hist read
  }

  if (t < NUM_EXPERTS) chunk_counts[blockIdx.x * NUM_EXPERTS + t] = hist[t];
}

__global__ __launch_bounds__(256)
void moe_prefix_kernel(int* __restrict__ chunk_counts,
                       int* __restrict__ expert_base,
                       float* __restrict__ out_counts,
                       int nchunks) {
  __shared__ int scn[256];
  const int e = threadIdx.x;
  int run = 0;
  for (int c = 0; c < nchunks; ++c) {
    int v = chunk_counts[c * NUM_EXPERTS + e];
    chunk_counts[c * NUM_EXPERTS + e] = run;   // per-expert, per-chunk exclusive prefix
    run += v;
  }
  out_counts[e] = (float)run;                  // num_tokens_per_expert
  scn[e] = run;
  __syncthreads();
  const int total = run;
  for (int off = 1; off < 256; off <<= 1) {    // Hillis-Steele inclusive scan
    int v = (e >= off) ? scn[e - off] : 0;
    __syncthreads();
    scn[e] += v;
    __syncthreads();
  }
  expert_base[e] = scn[e] - total;             // exclusive base per expert
}

__global__ __launch_bounds__(256)
void moe_scatter_kernel(const int*   __restrict__ topk_ids,
                        const float* __restrict__ topk_w,
                        const int*   __restrict__ chunk_counts,
                        const int*   __restrict__ expert_base,
                        float* __restrict__ out_scores,
                        float* __restrict__ out_tokidx) {
  __shared__ int   ids[CHUNK_ENTRIES];
  __shared__ float wts[CHUNK_ENTRIES];
  const int t = threadIdx.x;
  const int c = blockIdx.x;
  const int flatBase = c * CHUNK_ENTRIES;
  ids[t]       = topk_ids[flatBase + t];
  ids[t + 256] = topk_ids[flatBase + t + 256];
  wts[t]       = topk_w[flatBase + t];
  wts[t + 256] = topk_w[flatBase + t + 256];
  __syncthreads();
  // stable counting-sort scatter: thread t owns expert t
  int pos = expert_base[t] + chunk_counts[c * NUM_EXPERTS + t];
  for (int i = 0; i < CHUNK_ENTRIES; ++i) {
    if (ids[i] == t) {
      out_scores[pos] = wts[i];
      out_tokidx[pos] = (float)((flatBase + i) >> 3);  // token = flat_idx / EXPERTS_PER_TOKEN
      ++pos;
    }
  }
}

extern "C" void kernel_launch(void* const* d_in, const int* in_sizes, int n_in,
                              void* d_out, int out_size, void* d_ws, size_t ws_size,
                              hipStream_t stream) {
  const float* x    = (const float*)d_in[0];
  const float* gw   = (const float*)d_in[1];
  const float* bias = (const float*)d_in[2];
  float* out = (float*)d_out;  // [0,65536): scores_per_expert, [65536,131072): token idxs, [131072,131328): counts

  char* ws = (char*)d_ws;
  int*   topk_ids     = (int*)  (ws + 0);        // 65536 ints
  float* topk_w       = (float*)(ws + 262144);   // 65536 floats
  int*   chunk_counts = (int*)  (ws + 524288);   // 128*256 ints
  int*   expert_base  = (int*)  (ws + 655360);   // 256 ints

  moe_gemm_router_kernel<<<N_CHUNKS, THREADS, SM_BYTES, stream>>>(x, gw, bias, topk_ids, topk_w, chunk_counts);
  moe_prefix_kernel<<<1, 256, 0, stream>>>(chunk_counts, expert_base, out + 131072, N_CHUNKS);
  moe_scatter_kernel<<<N_CHUNKS, 256, 0, stream>>>(topk_ids, topk_w, chunk_counts, expert_base, out, out + 65536);
}
